// ActorCriticGNN_16819091931152
// MI455X (gfx1250) — matrix operand order repaired and checked by reading, not compiled, not run on hardware
//
#include <hip/hip_runtime.h>
#include <hip/hip_bf16.h>

#define NNODES 50000
#define NEDGES 1600000
#define F_IN   128
#define HID    256
#define ADIM   16
#define NGRAPH 64

typedef __attribute__((ext_vector_type(16))) __bf16 v16bf;
typedef __attribute__((ext_vector_type(8)))  float  v8f;

union ABfrag { v16bf v; unsigned short u[16]; uint4 q[2]; };

__device__ __forceinline__ unsigned int pack2bf(float a, float b) {
    union { __bf16 h[2]; unsigned int u; } r;
    r.h[0] = (__bf16)a;
    r.h[1] = (__bf16)b;
    return r.u;
}
__device__ __forceinline__ void unpack2bf(unsigned int u, float& a, float& b) {
    union { unsigned int u; __bf16 h[2]; } r;
    r.u = u;
    a = (float)r.h[0];
    b = (float)r.h[1];
}
__device__ __forceinline__ unsigned short bf16bits(float f) {
    union { __bf16 h; unsigned short s; } r;
    r.h = (__bf16)f;
    return r.s;
}

// ---------------------------------------------------------------- fill zeros
__global__ void fill_zero(float* __restrict__ p, size_t n4) {   // n4 = count/4
    size_t i = (size_t)blockIdx.x * blockDim.x + threadIdx.x;
    if (i < n4) ((float4*)p)[i] = make_float4(0.f, 0.f, 0.f, 0.f);
}

// ------------------------------------------------- edge scatter-add (mean aggr)
// one wave (32 lanes) per edge; lane handles F/32 consecutive features.
// BF16IN: features stored as bf16 (halves gather read traffic); accumulation
// stays f32 via global atomics.
template<int F, bool COUNT, bool BF16IN>
__global__ void scatter_accum(const void* __restrict__ featv,
                              const int*  __restrict__ srcArr,
                              const int*  __restrict__ dstArr,
                              float* __restrict__ agg,
                              float* __restrict__ cnt,
                              int nEdges) {
    int gtid = blockIdx.x * blockDim.x + threadIdx.x;
    int e    = gtid >> 5;
    int lane = gtid & 31;
    if (e >= nEdges) return;
    int s = srcArr[e];
    int d = dstArr[e];
    constexpr int PL = F / 32;                  // 4 (F=128) or 8 (F=256)
    float* ap = agg + (size_t)d * F + lane * PL;
    if constexpr (!BF16IN) {
        const float* fp = (const float*)featv + (size_t)s * F + lane * PL;
#pragma unroll
        for (int i = 0; i < PL; i += 4) {
            float4 v = *(const float4*)(fp + i);
            atomicAdd(ap + i + 0, v.x);
            atomicAdd(ap + i + 1, v.y);
            atomicAdd(ap + i + 2, v.z);
            atomicAdd(ap + i + 3, v.w);
        }
    } else {
        const unsigned short* fp = (const unsigned short*)featv + (size_t)s * F + lane * PL;
#pragma unroll
        for (int i = 0; i < PL; i += 8) {
            uint4 raw = *(const uint4*)(fp + i);        // 8 bf16
            float v0, v1, v2, v3, v4, v5, v6, v7;
            unpack2bf(raw.x, v0, v1);
            unpack2bf(raw.y, v2, v3);
            unpack2bf(raw.z, v4, v5);
            unpack2bf(raw.w, v6, v7);
            atomicAdd(ap + i + 0, v0); atomicAdd(ap + i + 1, v1);
            atomicAdd(ap + i + 2, v2); atomicAdd(ap + i + 3, v3);
            atomicAdd(ap + i + 4, v4); atomicAdd(ap + i + 5, v5);
            atomicAdd(ap + i + 6, v6); atomicAdd(ap + i + 7, v7);
        }
    }
    if (COUNT && lane == 0) atomicAdd(cnt + d, 1.0f);
}

// ---------------------------------------------------------------- SAGE GEMM
// out[N,256] = relu( (Aagg/cnt) @ Wl + Aself @ Wr + bias ), via bf16 WMMA.
// Block = 256 threads (8 wave32) -> 128-row x 64-col tile.
// LDS tiles in WMMA fragment order (2x ds_load_b128 per operand), double
// buffered: stage tile k+1 overlaps WMMAs of tile k, one barrier per K-step.
// SELF_BF16: Aself stored bf16 -> staging is a raw 8B copy (no cvt).
// OUT_BF16: store result as bf16 (b16 stores).
template<int F, bool SELF_BF16, bool OUT_BF16>
__global__ void sage_gemm_wmma(const float* __restrict__ Aagg,
                               const void*  __restrict__ Aself,
                               const float* __restrict__ cnt,
                               const float* __restrict__ Wl,
                               const float* __restrict__ Wr,
                               const float* __restrict__ bias,
                               void* __restrict__ out,
                               int nrows) {
    constexpr int LSTR = 24;                        // u16 per lane slot (16 data + 8 pad)
    __shared__ __align__(16) unsigned short lds_a[2][8 * 32 * LSTR];  // 2 x 12 KB
    __shared__ __align__(16) unsigned short lds_b[2][4 * 32 * LSTR];  // 2 x  6 KB
    __shared__ float rinv[128];

    const int tid     = threadIdx.x;
    const int lane    = tid & 31;
    const int w       = tid >> 5;         // wave id 0..7
    const int half    = lane >> 4;        // 0 / 1
    const int l16     = lane & 15;
    const int rowBase = blockIdx.x * 128;
    const int colBase = blockIdx.y * 64;

    if (tid < 128) {
        int row = rowBase + tid;
        float c = (row < nrows) ? cnt[row] : 1.0f;
        rinv[tid] = 1.0f / fmaxf(c, 1.0f);
    }

    auto stage = [&](int kt, unsigned short* la, unsigned short* lb) {
        const int  kbase  = kt * 32;
        const bool useAgg = (kbase < F);
        const float* Wsrc = useAgg ? Wl : Wr;
        const int    kk0  = useAgg ? kbase : (kbase - F);

        // A tile (128 rows x 32 k): k-quads. Quad (c..c+3, c%4==0) maps to 4
        // contiguous slots of one lane: lane = m + 16*((c>>3)&1),
        // slot j = (c&7) + ((c&16)?8:0).   128*8 = 1024 quads / 256 thr = 4 iters
#pragma unroll
        for (int p = tid; p < 128 * 8; p += 256) {
            int r = p >> 3;
            int c = (p & 7) * 4;
            int rowc = rowBase + r;
            rowc = (rowc < nrows) ? rowc : (nrows - 1);   // clamp: no exec divergence
            uint2 pk;
            if (useAgg) {
                float4 t4 = *(const float4*)(Aagg + (size_t)rowc * F + kk0 + c);
                float s = rinv[r];
                t4.x *= s; t4.y *= s; t4.z *= s; t4.w *= s;
                pk = make_uint2(pack2bf(t4.x, t4.y), pack2bf(t4.z, t4.w));
            } else if constexpr (SELF_BF16) {
                // bf16 source is already in (k,k+1) pair order: raw 8-byte copy
                pk = *(const uint2*)((const unsigned short*)Aself + (size_t)rowc * F + kk0 + c);
            } else {
                float4 t4 = *(const float4*)((const float*)Aself + (size_t)rowc * F + kk0 + c);
                pk = make_uint2(pack2bf(t4.x, t4.y), pack2bf(t4.z, t4.w));
            }
            int wv = r >> 4, m = r & 15;
            int h  = (c >> 3) & 1;
            int j  = (c & 7) + ((c & 16) ? 8 : 0);
            int ln = m + 16 * h;
            *(uint2*)(la + (wv * 32 + ln) * LSTR + j) = pk;
        }
        // B tile (32 k x 64 n): k-quads. lane = (n&15) + 16*((k>>3)&1), slot = (k&7) + ((k&16)?8:0).
        // 8 k-quads * 64 n = 512 quads / 256 thr = 2 iters
#pragma unroll
        for (int p = tid; p < 8 * 64; p += 256) {
            int kk = (p >> 6) * 4;
            int n  = p & 63;
            const float* wp = Wsrc + (size_t)(kk0 + kk) * HID + colBase + n;
            float b0 = wp[0 * HID];
            float b1 = wp[1 * HID];
            float b2 = wp[2 * HID];
            float b3 = wp[3 * HID];
            int ln = (n & 15) + 16 * ((kk >> 3) & 1);
            int nt = n >> 4;
            int j  = (kk & 7) + ((kk & 16) ? 8 : 0);
            uint2 pk = make_uint2(pack2bf(b0, b1), pack2bf(b2, b3));
            *(uint2*)(lb + (nt * 32 + ln) * LSTR + j) = pk;
        }
    };

    v8f acc[4];
#pragma unroll
    for (int nt = 0; nt < 4; ++nt) acc[nt] = (v8f){};

    const int KT = (2 * F) / 32;
    stage(0, lds_a[0], lds_b[0]);
    __syncthreads();

    for (int kt = 0; kt < KT; ++kt) {
        const int cur = kt & 1;
        if (kt + 1 < KT) stage(kt + 1, lds_a[cur ^ 1], lds_b[cur ^ 1]);

        ABfrag af;
        {
            const unsigned short* base = lds_a[cur] + (w * 32 + lane) * LSTR;
            af.q[0] = *(const uint4*)(base);
            af.q[1] = *(const uint4*)(base + 8);
        }
#pragma unroll
        for (int nt = 0; nt < 4; ++nt) {
            ABfrag bfr;
            const unsigned short* bb = lds_b[cur] + (nt * 32 + lane) * LSTR;
            bfr.q[0] = *(const uint4*)(bb);
            bfr.q[1] = *(const uint4*)(bb + 8);
            acc[nt] = __builtin_amdgcn_wmma_f32_16x16x32_bf16(
                false, af.v, false, bfr.v, (short)0, acc[nt], false, false);
        }
        __syncthreads();
    }

    // epilogue: bias + relu (C/D layout: VGPR v -> M = v + 8*half)
#pragma unroll
    for (int nt = 0; nt < 4; ++nt) {
        int col = colBase + nt * 16 + l16;
        float b = bias[col];
#pragma unroll
        for (int v = 0; v < 8; ++v) {
            int row = rowBase + w * 16 + v + 8 * half;
            if (row < nrows) {
                float val = fmaxf(acc[nt][v] + b, 0.0f);
                if constexpr (OUT_BF16) {
                    ((unsigned short*)out)[(size_t)row * HID + col] = bf16bits(val);
                } else {
                    ((float*)out)[(size_t)row * HID + col] = val;
                }
            }
        }
    }
}

// ---------------------------------------------------------------- global mean pool
__global__ void pool_scatter(const float* __restrict__ h,
                             const int*   __restrict__ batch,
                             float* __restrict__ pooled,
                             float* __restrict__ gcnt,
                             int nrows) {
    int gtid = blockIdx.x * blockDim.x + threadIdx.x;
    int n    = gtid >> 5;
    int lane = gtid & 31;
    if (n >= nrows) return;
    int b = batch[n];
    const float* hp = h      + (size_t)n * HID + lane * 8;
    float*       pp = pooled + (size_t)b * HID + lane * 8;
#pragma unroll
    for (int i = 0; i < 8; i += 4) {
        float4 v = *(const float4*)(hp + i);
        atomicAdd(pp + i + 0, v.x);
        atomicAdd(pp + i + 1, v.y);
        atomicAdd(pp + i + 2, v.z);
        atomicAdd(pp + i + 3, v.w);
    }
    if (lane == 0) atomicAdd(gcnt + b, 1.0f);
}

__global__ void graph_mean(const float* __restrict__ pooled,
                           const float* __restrict__ gcnt,
                           float* __restrict__ g) {
    int i = blockIdx.x * 256 + threadIdx.x;        // NGRAPH*HID threads
    g[i] = pooled[i] / fmaxf(gcnt[i >> 8], 1.0f);
}

// ---------------------------------------------------------------- actor/critic heads
__global__ void mlp_head(const float* __restrict__ g,
                         const float* __restrict__ Wa1, const float* __restrict__ ba1,
                         const float* __restrict__ Wa2, const float* __restrict__ ba2,
                         const float* __restrict__ Wc1, const float* __restrict__ bc1,
                         const float* __restrict__ Wc2, const float* __restrict__ bc2,
                         float* __restrict__ outp) {
    __shared__ float gv[HID], ha[HID], hc[HID];
    const int gid = blockIdx.x;
    const int t   = threadIdx.x;
    gv[t] = g[(size_t)gid * HID + t];
    __syncthreads();
    float sa = ba1[t], sc = bc1[t];
    for (int k = 0; k < HID; ++k) {
        float gk = gv[k];
        sa += gk * Wa1[(size_t)k * HID + t];
        sc += gk * Wc1[(size_t)k * HID + t];
    }
    ha[t] = fmaxf(sa, 0.0f);
    hc[t] = fmaxf(sc, 0.0f);
    __syncthreads();
    if (t < ADIM) {
        float s = ba2[t];
        for (int k = 0; k < HID; ++k) s += ha[k] * Wa2[(size_t)k * ADIM + t];
        outp[(size_t)gid * ADIM + t] = s;                       // mu
    }
    if (t == 0) {
        float s = bc2[0];
        for (int k = 0; k < HID; ++k) s += hc[k] * Wc2[k];
        outp[(size_t)NGRAPH * ADIM + gid] = s;                  // value
    }
}

// ---------------------------------------------------------------- launcher
extern "C" void kernel_launch(void* const* d_in, const int* in_sizes, int n_in,
                              void* d_out, int out_size, void* d_ws, size_t ws_size,
                              hipStream_t stream) {
    const float* x    = (const float*)d_in[0];
    const int*   ei   = (const int*)  d_in[1];
    const int*   srcA = ei;
    const int*   dstA = ei + NEDGES;
    const int*   batch= (const int*)  d_in[2];
    const float* Wl1  = (const float*)d_in[3];
    const float* bl1  = (const float*)d_in[4];
    const float* Wr1  = (const float*)d_in[5];
    const float* Wl2  = (const float*)d_in[6];
    const float* bl2  = (const float*)d_in[7];
    const float* Wr2  = (const float*)d_in[8];
    const float* Wa1  = (const float*)d_in[9];
    const float* ba1  = (const float*)d_in[10];
    const float* Wa2  = (const float*)d_in[11];
    const float* ba2  = (const float*)d_in[12];
    const float* Wc1  = (const float*)d_in[13];
    const float* bc1  = (const float*)d_in[14];
    const float* Wc2  = (const float*)d_in[15];
    const float* bc2  = (const float*)d_in[16];
    float* outp = (float*)d_out;

    // workspace carve-out (256B aligned); agg region (N*256 f32) also hosts agg1
    size_t off = 0;
    auto alloc = [&](size_t bytes) -> void* {
        size_t o = (off + 255) & ~size_t(255);
        off = o + bytes;
        return (void*)((char*)d_ws + o);
    };
    float*          cnt    = (float*)alloc((size_t)NNODES * 4);
    unsigned short* h1     = (unsigned short*)alloc((size_t)NNODES * HID * 2);  // bf16
    float*          agg    = (float*)alloc((size_t)NNODES * HID * 4);
    float*          h2     = (float*)alloc((size_t)NNODES * HID * 4);
    float*          pooled = (float*)alloc((size_t)NGRAPH * HID * 4);
    float*          gcnt   = (float*)alloc((size_t)NGRAPH * 4);
    float*          gvec   = (float*)alloc((size_t)NGRAPH * HID * 4);

    auto blocks4 = [](size_t n) { size_t n4 = n / 4; return (unsigned)((n4 + 255) / 256); };
    const unsigned edgeBlocks = (unsigned)(((size_t)NEDGES * 32 + 255) / 256);
    const unsigned nodeBlocks = (unsigned)(((size_t)NNODES * 32 + 255) / 256);
    const dim3 gemmGrid((NNODES + 127) / 128, HID / 64);

    // ---- layer 1: agg1 (f32 atomics) -> h1 (bf16) ----
    fill_zero<<<blocks4(NNODES), 256, 0, stream>>>(cnt, (size_t)NNODES / 4);
    fill_zero<<<blocks4((size_t)NNODES * F_IN), 256, 0, stream>>>(agg, (size_t)NNODES * F_IN / 4);
    scatter_accum<F_IN, true, false><<<edgeBlocks, 256, 0, stream>>>(x, srcA, dstA, agg, cnt, NEDGES);
    sage_gemm_wmma<F_IN, false, true><<<gemmGrid, 256, 0, stream>>>(
        agg, x, cnt, Wl1, Wr1, bl1, h1, NNODES);

    // ---- layer 2: gather h1 (bf16 reads, f32 atomic accum) -> h2 (f32) ----
    fill_zero<<<blocks4((size_t)NNODES * HID), 256, 0, stream>>>(agg, (size_t)NNODES * HID / 4);
    scatter_accum<HID, false, true><<<edgeBlocks, 256, 0, stream>>>(h1, srcA, dstA, agg, cnt, NEDGES);
    sage_gemm_wmma<HID, true, false><<<gemmGrid, 256, 0, stream>>>(
        agg, h1, cnt, Wl2, Wr2, bl2, h2, NNODES);

    // ---- global mean pool ----
    fill_zero<<<blocks4((size_t)NGRAPH * HID), 256, 0, stream>>>(pooled, (size_t)NGRAPH * HID / 4);
    fill_zero<<<1, 256, 0, stream>>>(gcnt, (size_t)NGRAPH / 4);
    pool_scatter<<<nodeBlocks, 256, 0, stream>>>(h2, batch, pooled, gcnt, NNODES);
    graph_mean<<<NGRAPH, 256, 0, stream>>>(pooled, gcnt, gvec);

    // ---- actor / critic heads ----
    mlp_head<<<NGRAPH, 256, 0, stream>>>(gvec, Wa1, ba1, Wa2, ba2, Wc1, bc1, Wc2, bc2, outp);
}
